// MambaBlock_87076166960151
// MI455X (gfx1250) — matrix operand order repaired and checked by reading, not compiled, not run on hardware
//
#include <hip/hip_runtime.h>
#include <hip/hip_bf16.h>
#include <cstdint>

typedef __attribute__((ext_vector_type(16))) __bf16 v16bf;
typedef __attribute__((ext_vector_type(8)))  float  v8f;

#define B_SZ   2
#define L_SZ   2048
#define H_SZ   1024
#define DIN    2048
#define DSTATE 16
#define DCONV  4
#define EPSF   1e-6f

// register blocking: each wave computes MT x NT tiles of 16x16
#define MT 4
#define NT 2

union Frag { v16bf v; uint4 q[2]; };

// ---------- bf16 helpers (round-to-nearest-even, NaN safe) ----------
__device__ __forceinline__ unsigned short f2bf_bits(float f) {
  union { float f; unsigned u; } v; v.f = f;
  unsigned u = v.u;
  if ((u & 0x7fffffffu) > 0x7f800000u) return (unsigned short)((u >> 16) | 0x0040u);
  unsigned r = u + 0x7fffu + ((u >> 16) & 1u);
  return (unsigned short)(r >> 16);
}
__device__ __forceinline__ float bf2f(unsigned short h) {
  union { unsigned u; float f; } v; v.u = ((unsigned)h) << 16;
  return v.f;
}

// ---------- RMSNorm -> bf16 activations ----------
__global__ void rmsnorm_bf16_kernel(const float* __restrict__ x,
                                    const float* __restrict__ w,
                                    unsigned short* __restrict__ hbf) {
  int t   = blockIdx.x;         // token 0..4095
  int tid = threadIdx.x;        // 256 threads
  __shared__ float red[256];
  const float* xr = x + (size_t)t * H_SZ;
  float s = 0.f;
  #pragma unroll
  for (int i = 0; i < H_SZ / 256; ++i) { float v = xr[tid + 256 * i]; s += v * v; }
  red[tid] = s; __syncthreads();
  for (int off = 128; off > 0; off >>= 1) {
    if (tid < off) red[tid] += red[tid + off];
    __syncthreads();
  }
  float scale = rsqrtf(red[0] / (float)H_SZ + EPSF);
  unsigned short* hr = hbf + (size_t)t * H_SZ;
  #pragma unroll
  for (int i = 0; i < H_SZ / 256; ++i) {
    int j = tid + 256 * i;
    hr[j] = f2bf_bits(xr[j] * scale * w[j]);
  }
}

// ---------- transpose f32 [K x N] -> bf16 [N x K] (weight prep) ----------
__global__ void transpose_to_bf16_kernel(const float* __restrict__ src,
                                         unsigned short* __restrict__ dst,
                                         int K, int N) {
  size_t idx = (size_t)blockIdx.x * blockDim.x + threadIdx.x;
  if (idx >= (size_t)K * N) return;
  int n = (int)(idx / K);
  int k = (int)(idx % K);
  dst[idx] = f2bf_bits(src[(size_t)k * N + n]);
}

// ---------- fragment load / mma helpers ----------
__device__ __forceinline__ void load_frags(const unsigned short* const* arow,
                                           const unsigned short* const* brow,
                                           int k0, Frag* fa, Frag* fb) {
  #pragma unroll
  for (int i = 0; i < MT; ++i) {
    fa[i].q[0] = *(const uint4*)(arow[i] + k0);
    fa[i].q[1] = *(const uint4*)(arow[i] + k0 + 16);
  }
  #pragma unroll
  for (int j = 0; j < NT; ++j) {
    fb[j].q[0] = *(const uint4*)(brow[j] + k0);
    fb[j].q[1] = *(const uint4*)(brow[j] + k0 + 16);
  }
}

__device__ __forceinline__ void mma_all(const Frag* fa, const Frag* fb, v8f acc[MT][NT]) {
  #pragma unroll
  for (int i = 0; i < MT; ++i)
    #pragma unroll
    for (int j = 0; j < NT; ++j)
      acc[i][j] = __builtin_amdgcn_wmma_f32_16x16x32_bf16(
          /*neg_a=*/false, fa[i].v, /*neg_b=*/false, fb[j].v,
          /*c_mod=*/(short)0, acc[i][j], /*reuse_a=*/false, /*reuse_b=*/false);
}

// ---------- register-blocked, double-buffered WMMA GEMM ----------
// A[MxK] bf16 row-major, Bw[NxK] bf16 (pre-transposed weights); K % 64 == 0.
// MODE 0: split columns -> xs / z (f32)
// MODE 1: split columns -> Bbuf / Cbuf / softplus -> delta
// MODE 2: out = resid + acc  (final projection + residual)
template <int MODE>
__global__ void __launch_bounds__(128, 1)
wmma_gemm_kernel(const unsigned short* __restrict__ A,
                 const unsigned short* __restrict__ Bw,
                 int M, int N, int K,
                 float* __restrict__ out0,
                 float* __restrict__ out1,
                 float* __restrict__ out2,
                 const float* __restrict__ resid) {
  const int lane = threadIdx.x & 31;
  const int wave = threadIdx.x >> 5;
  const int macro = blockIdx.x * (blockDim.x >> 5) + wave;
  const int tilesN = N / (NT * 16);
  const int tmac = macro / tilesN;
  const int tnac = macro - tmac * tilesN;
  if (tmac >= M / (MT * 16)) return;

  const int rm = tmac * (MT * 16);
  const int cn = tnac * (NT * 16);

  // A fragment: lane holds row m=lane&15; K-halves {0..7,16..23} or {8..15,24..31}
  const int abase_k = (lane >> 4) << 3;   // 0 or 8
  // B fragment: lane holds column n=lane&15; K-halves {0..7,16..23} or {8..15,24..31}
  const int bbase_k = (lane >> 4) << 3;   // 0 or 8

  const unsigned short* arow[MT];
  const unsigned short* brow[NT];
  #pragma unroll
  for (int i = 0; i < MT; ++i)
    arow[i] = A + (size_t)(rm + i * 16 + (lane & 15)) * K + abase_k;
  #pragma unroll
  for (int j = 0; j < NT; ++j)
    brow[j] = Bw + (size_t)(cn + j * 16 + (lane & 15)) * K + bbase_k;

  v8f acc[MT][NT];
  #pragma unroll
  for (int i = 0; i < MT; ++i)
    #pragma unroll
    for (int j = 0; j < NT; ++j)
      acc[i][j] = (v8f){0.f, 0.f, 0.f, 0.f, 0.f, 0.f, 0.f, 0.f};

  // double-buffered main loop: loads for step k+32 fly during WMMAs of step k
  Frag fa0[MT], fb0[NT], fa1[MT], fb1[NT];
  load_frags(arow, brow, 0, fa0, fb0);
  for (int k0 = 0; k0 < K; k0 += 64) {
    load_frags(arow, brow, k0 + 32, fa1, fb1);
    mma_all(fa0, fb0, acc);
    if (k0 + 64 < K) load_frags(arow, brow, k0 + 64, fa0, fb0);
    mma_all(fa1, fb1, acc);
  }

  // C/D layout: VGPR v holds row v (lanes 0-15) or v+8 (lanes 16-31), col = lane&15
  #pragma unroll
  for (int i = 0; i < MT; ++i) {
    const int rbase = rm + i * 16 + ((lane >> 4) << 3);
    #pragma unroll
    for (int j = 0; j < NT; ++j) {
      const int colg = cn + j * 16 + (lane & 15);
      #pragma unroll
      for (int v = 0; v < 8; ++v) {
        int rowg  = rbase + v;
        float val = acc[i][j][v];
        if (MODE == 0) {
          if (colg < DIN) out0[(size_t)rowg * DIN + colg] = val;
          else            out1[(size_t)rowg * DIN + (colg - DIN)] = val;
        } else if (MODE == 1) {
          if (colg < DSTATE)            out0[(size_t)rowg * DSTATE + colg] = val;
          else if (colg < 2 * DSTATE)   out1[(size_t)rowg * DSTATE + (colg - DSTATE)] = val;
          else {
            float sp = (val > 20.f) ? val : log1pf(__expf(val));
            out2[(size_t)rowg * DIN + (colg - 2 * DSTATE)] = sp;
          }
        } else {
          out0[(size_t)rowg * H_SZ + colg] = resid[(size_t)rowg * H_SZ + colg] + val;
        }
      }
    }
  }
}

// ---------- depthwise causal conv (k=4) + bias + SiLU -> bf16 ----------
__global__ void conv_silu_kernel(const float* __restrict__ xs,
                                 const float* __restrict__ cw,
                                 const float* __restrict__ cb,
                                 unsigned short* __restrict__ xsilu) {
  size_t idx = (size_t)blockIdx.x * blockDim.x + threadIdx.x;
  if (idx >= (size_t)B_SZ * L_SZ * DIN) return;
  int d     = (int)(idx % DIN);
  size_t bl = idx / DIN;          // token index b*L + l
  int l     = (int)(bl % L_SZ);
  float acc = cb[d];
  #pragma unroll
  for (int j = 0; j < DCONV; ++j) {
    int ls = l + j - (DCONV - 1);
    if (ls >= 0)
      acc += cw[d * DCONV + j] * xs[(bl + (size_t)(long)(j - (DCONV - 1))) * DIN + d];
  }
  float sig = 1.f / (1.f + __expf(-acc));
  xsilu[idx] = f2bf_bits(acc * sig);
}

// ---------- selective scan + D-skip + z-gating -> bf16 ----------
__global__ void scan_kernel(const float* __restrict__ Bbuf,
                            const float* __restrict__ Cbuf,
                            const float* __restrict__ delta,
                            const unsigned short* __restrict__ xsilu,
                            const float* __restrict__ zbuf,
                            const float* __restrict__ A_log,
                            const float* __restrict__ D_param,
                            unsigned short* __restrict__ ygate) {
  int d = blockIdx.x * blockDim.x + threadIdx.x;   // channel
  int b = blockIdx.y;                              // batch
  __shared__ float sB[DSTATE], sC[DSTATE];
  float Avals[DSTATE], h[DSTATE];
  #pragma unroll
  for (int n = 0; n < DSTATE; ++n) {
    Avals[n] = -__expf(A_log[d * DSTATE + n]);
    h[n] = 0.f;
  }
  float Dv = D_param[d];
  for (int l = 0; l < L_SZ; ++l) {
    size_t t = (size_t)b * L_SZ + l;
    if (threadIdx.x < DSTATE)
      sB[threadIdx.x] = Bbuf[t * DSTATE + threadIdx.x];
    else if (threadIdx.x < 2 * DSTATE)
      sC[threadIdx.x - DSTATE] = Cbuf[t * DSTATE + (threadIdx.x - DSTATE)];
    __syncthreads();
    float dt  = delta[t * DIN + d];
    float xv  = bf2f(xsilu[t * DIN + d]);
    float dtx = dt * xv;
    float y = 0.f;
    #pragma unroll
    for (int n = 0; n < DSTATE; ++n) {
      float dA = __expf(dt * Avals[n]);
      h[n] = dA * h[n] + dtx * sB[n];
      y += h[n] * sC[n];
    }
    y += Dv * xv;
    float zv = zbuf[t * DIN + d];
    y *= zv / (1.f + __expf(-zv));     // y * silu(z)
    ygate[t * DIN + d] = f2bf_bits(y);
    __syncthreads();
  }
}

// ---------- host launcher ----------
extern "C" void kernel_launch(void* const* d_in, const int* in_sizes, int n_in,
                              void* d_out, int out_size, void* d_ws, size_t ws_size,
                              hipStream_t stream) {
  (void)in_sizes; (void)n_in; (void)out_size; (void)ws_size;
  const float* x       = (const float*)d_in[0];
  const float* norm_w  = (const float*)d_in[1];
  const float* w_in    = (const float*)d_in[2];
  const float* conv_w  = (const float*)d_in[3];
  const float* conv_b  = (const float*)d_in[4];
  const float* w_xproj = (const float*)d_in[5];
  const float* A_log   = (const float*)d_in[6];
  const float* D_param = (const float*)d_in[7];
  const float* w_out   = (const float*)d_in[8];
  float* out = (float*)d_out;

  const int TOK = B_SZ * L_SZ;                  // 4096 tokens
  unsigned char* ws = (unsigned char*)d_ws;
  size_t off = 0;
  auto carve = [&](size_t bytes) -> void* {
    void* p = ws + off;
    off += (bytes + 255) & ~(size_t)255;
    return p;
  };
  unsigned short* hbf     = (unsigned short*)carve((size_t)TOK * H_SZ * 2);
  unsigned short* wA      = (unsigned short*)carve((size_t)H_SZ * (2 * DIN) * 2);
  float*          xs      = (float*)carve((size_t)TOK * DIN * 4);
  float*          zbuf    = (float*)carve((size_t)TOK * DIN * 4);
  unsigned short* xsiluBf = (unsigned short*)carve((size_t)TOK * DIN * 2);
  unsigned short* wX      = (unsigned short*)carve((size_t)(2 * DSTATE + DIN) * DIN * 2);
  float*          Bbuf    = (float*)carve((size_t)TOK * DSTATE * 4);
  float*          Cbuf    = (float*)carve((size_t)TOK * DSTATE * 4);
  float*          deltaB  = (float*)carve((size_t)TOK * DIN * 4);
  unsigned short* ygateBf = (unsigned short*)carve((size_t)TOK * DIN * 2);
  unsigned short* wO      = (unsigned short*)carve((size_t)H_SZ * DIN * 2);

  // 1) RMSNorm -> bf16 activations
  rmsnorm_bf16_kernel<<<TOK, 256, 0, stream>>>(x, norm_w, hbf);

  // 2) weight prep (transpose + bf16)
  {
    size_t tot = (size_t)H_SZ * (2 * DIN);
    transpose_to_bf16_kernel<<<(unsigned)((tot + 255) / 256), 256, 0, stream>>>(w_in, wA, H_SZ, 2 * DIN);
  }
  {
    size_t tot = (size_t)DIN * (2 * DSTATE + DIN);
    transpose_to_bf16_kernel<<<(unsigned)((tot + 255) / 256), 256, 0, stream>>>(w_xproj, wX, DIN, 2 * DSTATE + DIN);
  }
  {
    size_t tot = (size_t)DIN * H_SZ;
    transpose_to_bf16_kernel<<<(unsigned)((tot + 255) / 256), 256, 0, stream>>>(w_out, wO, DIN, H_SZ);
  }

  // 3) GEMM1: xz = h @ w_in  (split -> xs, z)
  {
    int M = TOK, N = 2 * DIN, K = H_SZ;
    int tiles = (M / (MT * 16)) * (N / (NT * 16));   // 64 * 128 = 8192
    wmma_gemm_kernel<0><<<tiles / 4, 128, 0, stream>>>(hbf, wA, M, N, K, xs, zbuf, nullptr, nullptr);
  }

  // 4) depthwise conv + SiLU
  {
    size_t tot = (size_t)TOK * DIN;
    conv_silu_kernel<<<(unsigned)((tot + 255) / 256), 256, 0, stream>>>(xs, conv_w, conv_b, xsiluBf);
  }

  // 5) GEMM2: proj = x_ssm @ w_xproj (split -> B, C, softplus delta)
  {
    int M = TOK, N = 2 * DSTATE + DIN, K = DIN;
    int tiles = (M / (MT * 16)) * (N / (NT * 16));   // 64 * 65 = 4160
    wmma_gemm_kernel<1><<<tiles / 4, 128, 0, stream>>>(xsiluBf, wX, M, N, K, Bbuf, Cbuf, deltaB, nullptr);
  }

  // 6) selective scan + D-skip + z-gate
  scan_kernel<<<dim3(DIN / 256, B_SZ), 256, 0, stream>>>(Bbuf, Cbuf, deltaB, xsiluBf,
                                                         zbuf, A_log, D_param, ygateBf);

  // 7) GEMM3: out = residual + y @ w_out
  {
    int M = TOK, N = H_SZ, K = DIN;
    int tiles = (M / (MT * 16)) * (N / (NT * 16));   // 64 * 32 = 2048
    wmma_gemm_kernel<2><<<tiles / 4, 128, 0, stream>>>(ygateBf, wO, M, N, K, out, nullptr, nullptr, x);
  }
}
